// MultiScaleSmallKernelConvAndAttention_53970559042337
// MI455X (gfx1250) — matrix operand order repaired and checked by reading, not compiled, not run on hardware
//
#include <hip/hip_runtime.h>

// ---------------------------------------------------------------------------
// MultiScaleSmallKernelConvAndAttention — MI455X (gfx1250, wave32, WMMA)
//
// x (4,64,96,96) fp32. Deformable convs ks=1,3,5 (offset conv -> bilinear
// gather -> kxk contraction), lrelu, summed; then 3-scale attention convs +
// sigmoid gate. All channel contractions run as V_WMMA_F32_16X16X32_F16 with
// fragment-layout operands: A fragments pre-swizzled in global memory (one
// 32B load per lane), B fragments staged in LDS in lane order (one 32B
// ds_load per lane).
// ---------------------------------------------------------------------------

typedef __attribute__((ext_vector_type(16))) _Float16 v16h;
typedef __attribute__((ext_vector_type(8)))  float    v8f;

constexpr int NB    = 4;
constexpr int NCH   = 64;
constexpr int IMG_H = 96;
constexpr int IMG_W = 96;
constexpr int HWP   = IMG_H * IMG_W;   // 9216
constexpr int PTOT  = NB * HWP;        // 36864
constexpr int POS_TILE = 64;           // positions per workgroup (64 | 9216)
constexpr int NBLK  = PTOT / POS_TILE; // 576, exact

static __device__ __forceinline__ v8f wmma_f16(v16h a, v16h b, v8f c) {
    return __builtin_amdgcn_wmma_f32_16x16x32_f16(false, a, false, b,
                                                  (short)0, c, false, false);
}

// ---------------------------------------------------------------------------
// Weight re-layout fp32 (O,C,ks,ks) -> f16 A-fragments:
//   dst[((n*2 + chunk)*4 + oct)*512 + lane*16 + s]
//   oc = oct*16 + (lane&15); hi = lane>>4
//   k  = chunk*32 + hi*16 + s   (contiguous k order, matches B staging)
// ---------------------------------------------------------------------------
__global__ void convert_w_kernel(const float* __restrict__ src,
                                 _Float16* __restrict__ dst,
                                 int N, int total) {
    int idx = blockIdx.x * blockDim.x + threadIdx.x;
    if (idx >= total) return;
    int s     = idx & 15;
    int lane  = (idx >> 4) & 31;
    int oct   = (idx >> 9) & 3;
    int chunk = (idx >> 11) & 1;
    int n     = idx >> 12;
    int oc = oct * 16 + (lane & 15);
    int hi = lane >> 4;
    int k  = chunk * 32 + hi * 16 + s;
    dst[idx] = (_Float16)src[(oc * 64 + k) * N + n];
}

__global__ void batt_kernel(const float* __restrict__ b1,
                            const float* __restrict__ b3,
                            const float* __restrict__ b5,
                            float* __restrict__ batt) {
    int i = threadIdx.x;
    if (i < NCH) batt[i] = b1[i] + b3[i] + b5[i];
}

// ---------------------------------------------------------------------------
// Offset conv: 3x3/pad1, C=64 -> 2N channels, fp32 VALU (too narrow for WMMA)
// ---------------------------------------------------------------------------
__global__ void offset_conv_kernel(const float* __restrict__ x,
                                   const float* __restrict__ wo,
                                   const float* __restrict__ bo,
                                   float* __restrict__ off, int CO) {
    int idx = blockIdx.x * blockDim.x + threadIdx.x;
    int total = NB * CO * HWP;
    if (idx >= total) return;
    int hw = idx % HWP;
    int t  = idx / HWP;
    int co = t % CO;
    int b  = t / CO;
    int h = hw / IMG_W, w = hw % IMG_W;
    float acc = bo[co];
    const float* xb = x + b * NCH * HWP;
    const float* wk = wo + co * NCH * 9;
    for (int ic = 0; ic < NCH; ++ic) {
        const float* xc = xb + ic * HWP;
        __builtin_prefetch(xc + hw + HWP, 0, 0);   // global_prefetch_b8
        const float* wc = wk + ic * 9;
#pragma unroll
        for (int di = -1; di <= 1; ++di) {
            int hh = h + di;
            if ((unsigned)hh >= (unsigned)IMG_H) continue;
#pragma unroll
            for (int dj = -1; dj <= 1; ++dj) {
                int ww = w + dj;
                if ((unsigned)ww >= (unsigned)IMG_W) continue;
                acc += xc[hh * IMG_W + ww] * wc[(di + 1) * 3 + (dj + 1)];
            }
        }
    }
    off[idx] = acc;
}

// ---------------------------------------------------------------------------
// Shared B-fragment LDS layout: XsB[((pt*2 + chunk)*32 + lane)*16 + s]
//   sample channel c = chunk*32 + (lane>>4)*16 + s, position = pt*16 + (lane&15)
// ---------------------------------------------------------------------------

// Deformable-conv GEMM. 8 waves: oct = wave&3 (oc tile), pth = wave>>2; each
// wave owns pos-subtiles 2*pth and 2*pth+1 (two accumulators, 4 WMMA / tap).
__global__ __launch_bounds__(256) void deform_gemm_kernel(
    const float* __restrict__ x, const float* __restrict__ off,
    const _Float16* __restrict__ wfrag, const float* __restrict__ bias,
    float* __restrict__ msf, int ksz, int pad, int accumulate) {

    const int N = ksz * ksz;
    __shared__ __align__(32) _Float16 XsB[4 * 2 * 32 * 16];  // 8 KB
    __shared__ float gw[4][POS_TILE];
    __shared__ int   gidx[4][POS_TILE];

    const int t = threadIdx.x;
    const int posbase = blockIdx.x * POS_TILE;
    const int batch  = posbase / HWP;    // uniform per block
    const int hwbase = posbase % HWP;    // uniform per block
    const int lane = t & 31;
    const int wave = t >> 5;
    const int oct  = wave & 3;
    const int pth  = wave >> 2;
    const int row  = lane & 15;
    const int hi   = lane >> 4;

    v8f acc0 = {}, acc1 = {};

    for (int n = 0; n < N; ++n) {
        // --- per-position bilinear coords/weights (threads 0..63) ---
        if (t < POS_TILE) {
            int hw = hwbase + t;
            int h = hw / IMG_W, w = hw % IMG_W;
            float offx = off[(batch * 2 * N + n) * HWP + hw];
            float offy = off[(batch * 2 * N + N + n) * HWP + hw];
            float px = offx + (float)(h + 1 + (n / ksz) - pad);
            float py = offy + (float)(w + 1 + (n % ksz) - pad);
            float Hm = (float)(IMG_H + 2 * pad - 1);
            float Wm = (float)(IMG_W + 2 * pad - 1);
            float fx = floorf(px), fy = floorf(py);
            float ltx = fminf(fmaxf(fx, 0.f), Hm);
            float lty = fminf(fmaxf(fy, 0.f), Wm);
            float rbx = fminf(fmaxf(fx + 1.f, 0.f), Hm);
            float rby = fminf(fmaxf(fy + 1.f, 0.f), Wm);
            float pxc = fminf(fmaxf(px, 0.f), Hm);
            float pyc = fminf(fmaxf(py, 0.f), Wm);
            float dx0 = 1.f + (ltx - pxc), dx1 = 1.f - (rbx - pxc);
            float dy0 = 1.f + (lty - pyc), dy1 = 1.f - (rby - pyc);
            gw[0][t] = dx0 * dy0;  gw[1][t] = dx1 * dy1;
            gw[2][t] = dx0 * dy1;  gw[3][t] = dx1 * dy0;
            int ix0 = (int)ltx - pad, iy0 = (int)lty - pad;
            int ix1 = (int)rbx - pad, iy1 = (int)rby - pad;
            auto lin = [](int qx, int qy) -> int {
                return (qx >= 0 && qx < IMG_H && qy >= 0 && qy < IMG_W)
                         ? qx * IMG_W + qy : -1;   // -1 => zero-pad region
            };
            gidx[0][t] = lin(ix0, iy0);
            gidx[1][t] = lin(ix1, iy1);
            gidx[2][t] = lin(ix0, iy1);
            gidx[3][t] = lin(ix1, iy0);
        }
        __syncthreads();

        // --- gather: 4 threads per position, 16 channels each, one 32B
        //     vectorized store straight into B-fragment layout ---
        {
            int lpos = t & 63;
            int c0   = (t >> 6) << 4;          // 0,16,32,48
            const float* xb = x + (batch * NCH + c0) * HWP;
            float w0 = gw[0][lpos], w1 = gw[1][lpos];
            float w2 = gw[2][lpos], w3 = gw[3][lpos];
            int i0 = gidx[0][lpos], i1 = gidx[1][lpos];
            int i2 = gidx[2][lpos], i3 = gidx[3][lpos];
            int chunk = c0 >> 5;
            int hiw   = (c0 >> 4) & 1;
            int pt    = lpos >> 4;
            int bl    = hiw * 16 + (lpos & 15);
            v16h vals;
#pragma unroll
            for (int cc = 0; cc < 16; ++cc) {
                const float* xc = xb + cc * HWP;
                float v = 0.f;
                if (i0 >= 0) v += w0 * xc[i0];
                if (i1 >= 0) v += w1 * xc[i1];
                if (i2 >= 0) v += w2 * xc[i2];
                if (i3 >= 0) v += w3 * xc[i3];
                vals[cc] = (_Float16)v;
            }
            *(v16h*)(XsB + ((pt * 2 + chunk) * 32 + bl) * 16) = vals;
        }
        __syncthreads();

        // --- WMMA: A fragments direct from global (pre-swizzled), B from LDS ---
        {
            const _Float16* wp = wfrag + ((n * 2 + 0) * 4 + oct) * 512 + lane * 16;
            v16h a0 = *(const v16h*)wp;            // chunk 0
            v16h a1 = *(const v16h*)(wp + 2048);   // chunk 1 (+4*512 halves)
            v16h b00 = *(const v16h*)(XsB + (((2 * pth + 0) * 2 + 0) * 32 + lane) * 16);
            v16h b01 = *(const v16h*)(XsB + (((2 * pth + 0) * 2 + 1) * 32 + lane) * 16);
            v16h b10 = *(const v16h*)(XsB + (((2 * pth + 1) * 2 + 0) * 32 + lane) * 16);
            v16h b11 = *(const v16h*)(XsB + (((2 * pth + 1) * 2 + 1) * 32 + lane) * 16);
            acc0 = wmma_f16(a0, b00, acc0);
            acc0 = wmma_f16(a1, b01, acc0);
            acc1 = wmma_f16(a0, b10, acc1);
            acc1 = wmma_f16(a1, b11, acc1);
        }
        __syncthreads();
    }

    // --- epilogue: bias + leaky relu, write/accumulate msf (2 tiles) ---
#pragma unroll
    for (int j = 0; j < 2; ++j) {
        v8f a = j ? acc1 : acc0;
        int hw = hwbase + (2 * pth + j) * 16 + row;
#pragma unroll
        for (int v = 0; v < 8; ++v) {
            int oc = oct * 16 + hi * 8 + v;     // D layout: VGPR v -> M = v + 8*hi
            float val = a[v] + bias[oc];
            val = val >= 0.f ? val : 0.1f * val;
            float* dst = msf + (batch * NCH + oc) * HWP + hw;
            if (accumulate) *dst += val;
            else            *dst  = val;
        }
    }
}

// ---------------------------------------------------------------------------
// Attention: fused 1x1 + 3x3 + 5x5 convs on msf (35 taps, K=64 each) as WMMA
// GEMM, then sigmoid gate: out = msf * (1 + sigmoid(g + bias_sum)).
// ---------------------------------------------------------------------------
__global__ __launch_bounds__(256) void attention_kernel(
    const float* __restrict__ msf, const _Float16* __restrict__ wfrag,
    const float* __restrict__ batt, float* __restrict__ out) {

    __shared__ __align__(32) _Float16 XsB[4 * 2 * 32 * 16];  // 8 KB

    const int t = threadIdx.x;
    const int posbase = blockIdx.x * POS_TILE;
    const int batch  = posbase / HWP;
    const int hwbase = posbase % HWP;
    const int lane = t & 31;
    const int wave = t >> 5;
    const int oct  = wave & 3;
    const int pth  = wave >> 2;
    const int row  = lane & 15;
    const int hi   = lane >> 4;

    v8f acc0 = {}, acc1 = {};

    for (int tap = 0; tap < 35; ++tap) {
        int ksz, pad, n;
        if (tap == 0)       { ksz = 1; pad = 0; n = 0; }
        else if (tap < 10)  { ksz = 3; pad = 1; n = tap - 1; }
        else                { ksz = 5; pad = 2; n = tap - 10; }
        int di = n / ksz - pad, dj = n % ksz - pad;

        // shifted gather from msf straight into B-fragment layout
        {
            int lpos = t & 63;
            int c0   = (t >> 6) << 4;
            int hw = hwbase + lpos;
            int h = hw / IMG_W, w = hw % IMG_W;
            int hh = h + di, ww = w + dj;
            bool ok = (unsigned)hh < (unsigned)IMG_H && (unsigned)ww < (unsigned)IMG_W;
            const float* mb = msf + (batch * NCH + c0) * HWP
                                  + (ok ? hh * IMG_W + ww : 0);
            int chunk = c0 >> 5;
            int hiw   = (c0 >> 4) & 1;
            int pt    = lpos >> 4;
            int bl    = hiw * 16 + (lpos & 15);
            v16h vals;
#pragma unroll
            for (int cc = 0; cc < 16; ++cc)
                vals[cc] = ok ? (_Float16)mb[cc * HWP] : (_Float16)0.f;
            *(v16h*)(XsB + ((pt * 2 + chunk) * 32 + bl) * 16) = vals;
        }
        __syncthreads();

        {
            const _Float16* wp = wfrag + ((tap * 2 + 0) * 4 + oct) * 512 + lane * 16;
            v16h a0 = *(const v16h*)wp;
            v16h a1 = *(const v16h*)(wp + 2048);
            v16h b00 = *(const v16h*)(XsB + (((2 * pth + 0) * 2 + 0) * 32 + lane) * 16);
            v16h b01 = *(const v16h*)(XsB + (((2 * pth + 0) * 2 + 1) * 32 + lane) * 16);
            v16h b10 = *(const v16h*)(XsB + (((2 * pth + 1) * 2 + 0) * 32 + lane) * 16);
            v16h b11 = *(const v16h*)(XsB + (((2 * pth + 1) * 2 + 1) * 32 + lane) * 16);
            acc0 = wmma_f16(a0, b00, acc0);
            acc0 = wmma_f16(a1, b01, acc0);
            acc1 = wmma_f16(a0, b10, acc1);
            acc1 = wmma_f16(a1, b11, acc1);
        }
        __syncthreads();
    }

#pragma unroll
    for (int j = 0; j < 2; ++j) {
        v8f a = j ? acc1 : acc0;
        int hw = hwbase + (2 * pth + j) * 16 + row;
#pragma unroll
        for (int v = 0; v < 8; ++v) {
            int oc = oct * 16 + hi * 8 + v;
            float g = a[v] + batt[oc];
            float att = 1.f / (1.f + expf(-g));
            float m = msf[(batch * NCH + oc) * HWP + hw];
            out[(batch * NCH + oc) * HWP + hw] = m * (1.f + att);
        }
    }
}

// ---------------------------------------------------------------------------
// Host launch
// ---------------------------------------------------------------------------
extern "C" void kernel_launch(void* const* d_in, const int* in_sizes, int n_in,
                              void* d_out, int out_size, void* d_ws, size_t ws_size,
                              hipStream_t stream) {
    const float* x   = (const float*)d_in[0];
    const float* wo1 = (const float*)d_in[1];
    const float* bo1 = (const float*)d_in[2];
    const float* w1  = (const float*)d_in[3];
    const float* b1  = (const float*)d_in[4];
    const float* wa1 = (const float*)d_in[5];
    const float* ba1 = (const float*)d_in[6];
    const float* wo3 = (const float*)d_in[7];
    const float* bo3 = (const float*)d_in[8];
    const float* w3  = (const float*)d_in[9];
    const float* b3  = (const float*)d_in[10];
    const float* wa3 = (const float*)d_in[11];
    const float* ba3 = (const float*)d_in[12];
    const float* wo5 = (const float*)d_in[13];
    const float* bo5 = (const float*)d_in[14];
    const float* w5  = (const float*)d_in[15];
    const float* b5  = (const float*)d_in[16];
    const float* wa5 = (const float*)d_in[17];
    const float* ba5 = (const float*)d_in[18];

    // workspace carve-up (floats first, then 32B-aligned f16 fragment region)
    float* off1 = (float*)d_ws;                 // 4*2*9216
    float* off3 = off1 + NB * 2  * HWP;         // 4*18*9216
    float* off5 = off3 + NB * 18 * HWP;         // 4*50*9216
    float* msf  = off5 + NB * 50 * HWP;         // 4*64*9216
    float* batt = msf  + NB * NCH * HWP;        // 64
    _Float16* w1h = (_Float16*)(batt + 64);     // 4096 halves
    _Float16* w3h = w1h + 4096;                 // 36864
    _Float16* w5h = w3h + 36864;                // 102400
    _Float16* wah = w5h + 102400;               // 35*4096

    // 1) weight fp32->f16 A-fragment re-layout
    convert_w_kernel<<<16,  256, 0, stream>>>(w1,  w1h, 1,  4096);
    convert_w_kernel<<<144, 256, 0, stream>>>(w3,  w3h, 9,  36864);
    convert_w_kernel<<<400, 256, 0, stream>>>(w5,  w5h, 25, 102400);
    convert_w_kernel<<<16,  256, 0, stream>>>(wa1, wah,            1,  4096);
    convert_w_kernel<<<144, 256, 0, stream>>>(wa3, wah + 4096,     9,  36864);
    convert_w_kernel<<<400, 256, 0, stream>>>(wa5, wah + 10*4096, 25, 102400);
    batt_kernel<<<1, 64, 0, stream>>>(ba1, ba3, ba5, batt);

    // 2) offset convs (fp32 direct)
    offset_conv_kernel<<<(NB*2*HWP  + 255)/256, 256, 0, stream>>>(x, wo1, bo1, off1, 2);
    offset_conv_kernel<<<(NB*18*HWP + 255)/256, 256, 0, stream>>>(x, wo3, bo3, off3, 18);
    offset_conv_kernel<<<(NB*50*HWP + 255)/256, 256, 0, stream>>>(x, wo5, bo5, off5, 50);

    // 3) deformable WMMA GEMMs -> msf (first writes, rest accumulate)
    deform_gemm_kernel<<<NBLK, 256, 0, stream>>>(x, off1, w1h, b1, msf, 1, 0, 0);
    deform_gemm_kernel<<<NBLK, 256, 0, stream>>>(x, off3, w3h, b3, msf, 3, 1, 1);
    deform_gemm_kernel<<<NBLK, 256, 0, stream>>>(x, off5, w5h, b5, msf, 5, 2, 1);

    // 4) fused attention + gating -> d_out
    attention_kernel<<<NBLK, 256, 0, stream>>>(msf, wah, batt, (float*)d_out);
}
